// JoraLayer_21337397526537
// MI455X (gfx1250) — compile-verified, matrix operand order checked by reading, not run on hardware
//
#include <hip/hip_runtime.h>
#include <cstdint>

// ---------------------------------------------------------------------------
// Types for WMMA (CDNA5 / gfx1250, wave32)
// ---------------------------------------------------------------------------
typedef __attribute__((ext_vector_type(8)))  float  v8f;
typedef __attribute__((ext_vector_type(16))) __bf16 v16bf;
typedef __attribute__((ext_vector_type(8)))  __bf16 v8bf;
typedef __attribute__((ext_vector_type(4)))  int    v4i;

union V16 { v16bf v; v8bf h[2]; };

#define EPS 1e-8f

// gfx1250 async global->LDS copy path (ASYNCcnt-tracked).
#if defined(__has_builtin)
#  if __has_builtin(__builtin_amdgcn_global_load_async_to_lds_b128)
#    define HAVE_ASYNC_COPY 1
#  endif
#endif

#ifdef HAVE_ASYNC_COPY
// Builtin prototype (from the compiler diagnostic): first arg is a NON-const
// v4i pointer in the __device__ (AS1) space; second is the LDS (AS3) v4i ptr.
// Address-space changes must go through integer casts in HIP C++.
#  define AS1P(p) ((__attribute__((address_space(1))) v4i*)(uintptr_t)(p))
#  define AS3P(p) ((__attribute__((address_space(3))) v4i*)(uint32_t)(uintptr_t)(p))
#  if __has_builtin(__builtin_amdgcn_s_wait_asynccnt)
#    define WAIT_ASYNC() __builtin_amdgcn_s_wait_asynccnt(0)
#  else
#    define WAIT_ASYNC() asm volatile("s_wait_asynccnt 0x0" ::: "memory")
#  endif
#endif

// ---------------------------------------------------------------------------
// 1) Rotation maps: for each row r:  Wp1[r] = aL[r]*W[r] + bL[r]*W[rp[r]]
//    for each col c:  Wp[:,c] = aC[c]*Wp1[:,c] + bC[c]*Wp1[:,cp[c]]
// ---------------------------------------------------------------------------
__global__ void init_maps_kernel(float* aL, float* bL, int* rp,
                                 float* aC, float* bC, int* cp,
                                 int N, int M) {
  int i = blockIdx.x * blockDim.x + threadIdx.x;
  if (i < N) { aL[i] = 1.f; bL[i] = 0.f; rp[i] = i; }
  if (i < M) { aC[i] = 1.f; bC[i] = 0.f; cp[i] = i; }
}

__global__ void set_maps_kernel(const float* __restrict__ thL, const int* __restrict__ pL,
                                const float* __restrict__ thR, const int* __restrict__ pR,
                                float* aL, float* bL, int* rp,
                                float* aC, float* bC, int* cp,
                                int SL, int SR) {
  int s = blockIdx.x * blockDim.x + threadIdx.x;
  if (s < SL) {
    float c = cosf(thL[s]), sn = sinf(thL[s]);
    int i = pL[2*s], j = pL[2*s+1];
    aL[i] = c; bL[i] = -sn; rp[i] = j;   // row i = c*ri - s*rj
    aL[j] = c; bL[j] =  sn; rp[j] = i;   // row j = s*ri + c*rj
  }
  if (s < SR) {
    float c = cosf(thR[s]), sn = sinf(thR[s]);
    int p = pR[2*s], q = pR[2*s+1];
    aC[p] = c; bC[p] = -sn; cp[p] = q;   // col p = c*cp - s*cq
    aC[q] = c; bC[q] =  sn; cp[q] = p;   // col q = s*cp + c*cq
  }
}

// ---------------------------------------------------------------------------
// 2) Base row norms of the ORIGINAL W
// ---------------------------------------------------------------------------
__global__ __launch_bounds__(256) void base_norm_kernel(const float* __restrict__ W,
                                                        float* __restrict__ bn, int M) {
  __shared__ float red[256];
  int r = blockIdx.x;
  const float* Wr = W + (size_t)r * M;
  float ss = 0.f;
  for (int c = threadIdx.x; c < M; c += 256) { float w = Wr[c]; ss += w * w; }
  red[threadIdx.x] = ss; __syncthreads();
  for (int s = 128; s > 0; s >>= 1) {
    if (threadIdx.x < s) red[threadIdx.x] += red[threadIdx.x + s];
    __syncthreads();
  }
  if (threadIdx.x == 0) bn[r] = sqrtf(red[0]);
}

// ---------------------------------------------------------------------------
// 3) Fused rotate (left+right) + bf16 hi/lo split + rotated row sum-of-squares
// ---------------------------------------------------------------------------
__global__ __launch_bounds__(256) void rotate_split_kernel(
    const float* __restrict__ W,
    const float* __restrict__ aL, const float* __restrict__ bL, const int* __restrict__ rp,
    const float* __restrict__ aC, const float* __restrict__ bC, const int* __restrict__ cp,
    __bf16* __restrict__ whi, __bf16* __restrict__ wlo,
    float* __restrict__ rn2, int M) {
  __shared__ float red[256];
  int r = blockIdx.x;
  float al = aL[r], bl = bL[r];
  int pr = rp[r];
  const float* Wr = W + (size_t)r  * M;
  const float* Wq = W + (size_t)pr * M;
  __bf16* hr = whi + (size_t)r * M;
  __bf16* lr = wlo + (size_t)r * M;
  float ss = 0.f;
  for (int c = threadIdx.x; c < M; c += 256) {
    int q = cp[c];
    float w1 = al * Wr[c] + bl * Wq[c];   // left-rotated, this column
    float w2 = al * Wr[q] + bl * Wq[q];   // left-rotated, partner column
    float wp = aC[c] * w1 + bC[c] * w2;   // right rotation
    ss += wp * wp;
    __bf16 h = (__bf16)wp;
    hr[c] = h;
    lr[c] = (__bf16)(wp - (float)h);
  }
  red[threadIdx.x] = ss; __syncthreads();
  for (int s = 128; s > 0; s >>= 1) {
    if (threadIdx.x < s) red[threadIdx.x] += red[threadIdx.x + s];
    __syncthreads();
  }
  if (threadIdx.x == 0) rn2[r] = red[0];
}

// ---------------------------------------------------------------------------
// 4) Per-output-column scale = base_norm * exp(ecd) / (||Wp_row|| + eps)
// ---------------------------------------------------------------------------
__global__ void scale_kernel(const float* __restrict__ bn, const float* __restrict__ rn2,
                             const float* __restrict__ ecd, float* __restrict__ scale, int N) {
  int n = blockIdx.x * blockDim.x + threadIdx.x;
  if (n < N) scale[n] = bn[n] * expf(ecd[n]) / (sqrtf(rn2[n]) + EPS);
}

// ---------------------------------------------------------------------------
// 5) Split activations x into bf16 hi/lo
// ---------------------------------------------------------------------------
__global__ void split_x_kernel(const float* __restrict__ x,
                               __bf16* __restrict__ xhi, __bf16* __restrict__ xlo,
                               size_t total) {
  size_t i = (size_t)blockIdx.x * blockDim.x + threadIdx.x;
  if (i < total) {
    float v = x[i];
    __bf16 h = (__bf16)v;
    xhi[i] = h;
    xlo[i] = (__bf16)(v - (float)h);
  }
}

// ---------------------------------------------------------------------------
// 6) bf16x3 WMMA GEMM with DOUBLE-BUFFERED LDS + async global->LDS copies:
//    out[t,n] = scale[n] * (x[t,:] . Wp[n,:]) + bias[n]
//    Block tile 128(M) x 128(N) x 32(K); 8 waves arranged 2(M) x 4(N);
//    each wave owns 4x2 16x16 accum tiles -> 24 v_wmma per wave per K-step.
//    One barrier per K-step; next tile's copy overlaps current tile's WMMAs.
// ---------------------------------------------------------------------------
#define BM 128
#define BN 128
#define BK 32
#define LDT 40   // LDS row stride in bf16 elements (80 B: 16-B aligned, bank-skewed)

__global__ __launch_bounds__(256) void gemm_bf16x3_kernel(
    const __bf16* __restrict__ xhi, const __bf16* __restrict__ xlo,
    const __bf16* __restrict__ whi, const __bf16* __restrict__ wlo,
    const float* __restrict__ scale, const float* __restrict__ bias,
    float* __restrict__ out, int N, int K) {
  __shared__ __bf16 sAhi[2][BM * LDT], sAlo[2][BM * LDT];
  __shared__ __bf16 sBhi[2][BN * LDT], sBlo[2][BN * LDT];

  const int tid  = threadIdx.x;
  const int lane = tid & 31;
  const int wave = tid >> 5;
  const int wm   = wave & 1;    // wave row (M)
  const int wn   = wave >> 1;   // wave col (N)
  const int l16  = lane & 15;
  const int lh   = lane >> 4;
  const int m0   = blockIdx.y * BM;
  const int n0   = blockIdx.x * BN;

  // Cooperative-copy geometry: 512 16-B chunks per matrix, 2 per thread.
  // Chunk r of this thread: row = row0 + 64*r, element offset ch0 within BK.
  const int row0 = tid >> 2;
  const int ch0  = (tid & 3) << 3;

  v8f zero = {};
  v8f acc[4][2];
#pragma unroll
  for (int mt = 0; mt < 4; ++mt)
#pragma unroll
    for (int nt = 0; nt < 2; ++nt) acc[mt][nt] = zero;

#ifdef HAVE_ASYNC_COPY
  // ---- gfx1250 async global->LDS staging (ASYNCcnt) ----
  auto astage = [&](int buf, int k0) {
#pragma unroll
    for (int r = 0; r < 2; ++r) {
      size_t ga = (size_t)(m0 + row0 + 64*r) * K + k0 + ch0;
      size_t gb = (size_t)(n0 + row0 + 64*r) * K + k0 + ch0;
      int lo = (row0 + 64*r) * LDT + ch0;
      __builtin_amdgcn_global_load_async_to_lds_b128(AS1P(&xhi[ga]), AS3P(&sAhi[buf][lo]), 0, 0);
      __builtin_amdgcn_global_load_async_to_lds_b128(AS1P(&xlo[ga]), AS3P(&sAlo[buf][lo]), 0, 0);
      __builtin_amdgcn_global_load_async_to_lds_b128(AS1P(&whi[gb]), AS3P(&sBhi[buf][lo]), 0, 0);
      __builtin_amdgcn_global_load_async_to_lds_b128(AS1P(&wlo[gb]), AS3P(&sBlo[buf][lo]), 0, 0);
    }
  };
#else
  // ---- fallback: software-pipelined synchronous staging ----
  v8bf rAh[2], rAl[2], rBh[2], rBl[2];
  auto gload = [&](int k0) {
#pragma unroll
    for (int r = 0; r < 2; ++r) {
      size_t ga = (size_t)(m0 + row0 + 64*r) * K + k0 + ch0;
      size_t gb = (size_t)(n0 + row0 + 64*r) * K + k0 + ch0;
      rAh[r] = *(const v8bf*)&xhi[ga];
      rAl[r] = *(const v8bf*)&xlo[ga];
      rBh[r] = *(const v8bf*)&whi[gb];
      rBl[r] = *(const v8bf*)&wlo[gb];
    }
  };
  auto sstore = [&](int buf) {
#pragma unroll
    for (int r = 0; r < 2; ++r) {
      int lo = (row0 + 64*r) * LDT + ch0;
      *(v8bf*)&sAhi[buf][lo] = rAh[r];
      *(v8bf*)&sAlo[buf][lo] = rAl[r];
      *(v8bf*)&sBhi[buf][lo] = rBh[r];
      *(v8bf*)&sBlo[buf][lo] = rBl[r];
    }
  };
#endif

  // Fragment loads follow the documented 16-bit A layout: lanes 0-15 hold
  // K=0..7 & 16..23, lanes 16-31 hold K=8..15 & 24..31.
  auto compute = [&](int buf) {
    V16 fahi[4], falo[4], fbhi[2], fblo[2];
#pragma unroll
    for (int mt = 0; mt < 4; ++mt) {
      int base = ((wm * 4 + mt) * 16 + l16) * LDT + lh * 8;
      fahi[mt].h[0] = *(const v8bf*)&sAhi[buf][base];
      fahi[mt].h[1] = *(const v8bf*)&sAhi[buf][base + 16];
      falo[mt].h[0] = *(const v8bf*)&sAlo[buf][base];
      falo[mt].h[1] = *(const v8bf*)&sAlo[buf][base + 16];
    }
#pragma unroll
    for (int nt = 0; nt < 2; ++nt) {
      int base = ((wn * 2 + nt) * 16 + l16) * LDT + lh * 8;
      fbhi[nt].h[0] = *(const v8bf*)&sBhi[buf][base];
      fbhi[nt].h[1] = *(const v8bf*)&sBhi[buf][base + 16];
      fblo[nt].h[0] = *(const v8bf*)&sBlo[buf][base];
      fblo[nt].h[1] = *(const v8bf*)&sBlo[buf][base + 16];
    }
    // bf16x3 fp32 emulation: hi*hi + hi*lo + lo*hi
#pragma unroll
    for (int mt = 0; mt < 4; ++mt)
#pragma unroll
      for (int nt = 0; nt < 2; ++nt) {
        acc[mt][nt] = __builtin_amdgcn_wmma_f32_16x16x32_bf16(
            false, fahi[mt].v, false, fbhi[nt].v, (short)0, acc[mt][nt], false, false);
        acc[mt][nt] = __builtin_amdgcn_wmma_f32_16x16x32_bf16(
            false, fahi[mt].v, false, fblo[nt].v, (short)0, acc[mt][nt], false, false);
        acc[mt][nt] = __builtin_amdgcn_wmma_f32_16x16x32_bf16(
            false, falo[mt].v, false, fbhi[nt].v, (short)0, acc[mt][nt], false, false);
      }
  };

#ifdef HAVE_ASYNC_COPY
  astage(0, 0);
  WAIT_ASYNC();
  __syncthreads();
  int buf = 0;
  for (int k0 = 0; k0 < K; k0 += BK) {
    if (k0 + BK < K) astage(buf ^ 1, k0 + BK);  // copy next tile during WMMAs
    compute(buf);
    WAIT_ASYNC();                                // this wave's async copies done
    __syncthreads();                             // all waves' copies done
    buf ^= 1;
  }
#else
  gload(0);
  int buf = 0;
  for (int k0 = 0; k0 < K; k0 += BK) {
    sstore(buf);
    __syncthreads();
    if (k0 + BK < K) {
      gload(k0 + BK);                            // issue next tile's global loads
      __builtin_prefetch(&xhi[(size_t)(m0 + row0) * K + k0 + 2 * BK], 0, 0);
      __builtin_prefetch(&whi[(size_t)(n0 + row0) * K + k0 + 2 * BK], 0, 0);
    }
    compute(buf);                                // WMMAs overlap in-flight loads
    buf ^= 1;
  }
#endif

  // ---- epilogue: C layout = VGPR v, lanes 0-15: M=v,N=lane; lanes 16-31: M=v+8 ----
#pragma unroll
  for (int nt = 0; nt < 2; ++nt) {
    int n = n0 + (wn * 2 + nt) * 16 + l16;
    float sc = scale[n];
    float bi = bias[n];
#pragma unroll
    for (int mt = 0; mt < 4; ++mt) {
      int tbase = m0 + (wm * 4 + mt) * 16 + lh * 8;
#pragma unroll
      for (int v = 0; v < 8; ++v)
        out[(size_t)(tbase + v) * N + n] = acc[mt][nt][v] * sc + bi;
    }
  }
}

// ---------------------------------------------------------------------------
// Host-side launch
// ---------------------------------------------------------------------------
extern "C" void kernel_launch(void* const* d_in, const int* in_sizes, int n_in,
                              void* d_out, int out_size, void* d_ws, size_t ws_size,
                              hipStream_t stream) {
  const float* x    = (const float*)d_in[0];
  const float* W    = (const float*)d_in[1];
  const float* bias = (const float*)d_in[2];
  const float* thL  = (const float*)d_in[3];
  const float* thR  = (const float*)d_in[4];
  const float* ecd  = (const float*)d_in[5];
  const int*   pL   = (const int*)d_in[6];
  const int*   pR   = (const int*)d_in[7];

  const int N  = in_sizes[2];          // 4096 output features
  const int SL = in_sizes[3];          // 2048 left rotations
  const int SR = in_sizes[4];          // 2048 right rotations
  const int M  = in_sizes[1] / N;      // 4096 reduction dim
  const int T  = in_sizes[0] / M;      // 2048 tokens

  float* out = (float*)d_out;

  // Workspace carve-up (256-B aligned slices)
  char* ws = (char*)d_ws;
  size_t off = 0;
  auto take = [&](size_t bytes) -> char* {
    char* p = ws + off;
    off = (off + bytes + 255) & ~(size_t)255;
    return p;
  };
  __bf16* whi = (__bf16*)take((size_t)N * M * sizeof(__bf16));
  __bf16* wlo = (__bf16*)take((size_t)N * M * sizeof(__bf16));
  __bf16* xhi = (__bf16*)take((size_t)T * M * sizeof(__bf16));
  __bf16* xlo = (__bf16*)take((size_t)T * M * sizeof(__bf16));
  float*  bn    = (float*)take((size_t)N * sizeof(float));
  float*  rn2   = (float*)take((size_t)N * sizeof(float));
  float*  scale = (float*)take((size_t)N * sizeof(float));
  float*  aL    = (float*)take((size_t)N * sizeof(float));
  float*  bL    = (float*)take((size_t)N * sizeof(float));
  int*    rp    = (int*)  take((size_t)N * sizeof(int));
  float*  aC    = (float*)take((size_t)M * sizeof(float));
  float*  bC    = (float*)take((size_t)M * sizeof(float));
  int*    cp    = (int*)  take((size_t)M * sizeof(int));

  int NM = (N > M) ? N : M;
  init_maps_kernel<<<(NM + 255) / 256, 256, 0, stream>>>(aL, bL, rp, aC, bC, cp, N, M);
  int Smax = (SL > SR) ? SL : SR;
  set_maps_kernel<<<(Smax + 255) / 256, 256, 0, stream>>>(thL, pL, thR, pR,
                                                          aL, bL, rp, aC, bC, cp, SL, SR);
  base_norm_kernel<<<N, 256, 0, stream>>>(W, bn, M);
  rotate_split_kernel<<<N, 256, 0, stream>>>(W, aL, bL, rp, aC, bC, cp, whi, wlo, rn2, M);
  scale_kernel<<<(N + 255) / 256, 256, 0, stream>>>(bn, rn2, ecd, scale, N);
  size_t totalx = (size_t)T * M;
  split_x_kernel<<<(unsigned)((totalx + 255) / 256), 256, 0, stream>>>(x, xhi, xlo, totalx);

  dim3 grid(N / BN, T / BM);
  gemm_bf16x3_kernel<<<grid, 256, 0, stream>>>(xhi, xlo, whi, wlo, scale, bias, out, N, M);
}
